// DCNv4PackFlowGuided_36017595744459
// MI455X (gfx1250) — matrix operand order repaired and checked by reading, not compiled, not run on hardware
//
#include <hip/hip_runtime.h>
#include <stdint.h>
#include <stddef.h>

// ---------------------------------------------------------------------------
// Types for CDNA5 WMMA (wave32): v_wmma_f32_16x16x32_bf16
// ---------------------------------------------------------------------------
typedef __attribute__((ext_vector_type(8)))  __bf16 v8bf;
typedef __attribute__((ext_vector_type(16))) __bf16 v16bf;
typedef __attribute__((ext_vector_type(8)))  float  v8f;
typedef uint32_t u32x4 __attribute__((ext_vector_type(4)));
typedef int      i32x8 __attribute__((ext_vector_type(8)));
typedef int      i32x4 __attribute__((ext_vector_type(4)));

#define NB  4      // batch
#define NC  64     // channels
#define NH  256
#define NW  256
#define NHP 258    // padded spatial (halo 1)
#define NWP 258
#define NG  4      // deformable groups
#define NCG 16     // channels per group
#define OMC 112    // DCN_OC (offset/mask channels)
#define OMINC 130  // conv1 input channels (2*64+2)
#define OMINCP 160 // conv1 input channels padded to multiple of 32

#if __has_builtin(__builtin_amdgcn_tensor_load_to_lds) && \
    __has_builtin(__builtin_amdgcn_s_wait_tensorcnt)
#define USE_TDM 1
#else
#define USE_TDM 0
#endif

__device__ __forceinline__ uint16_t f2bf(float f) {
  // round-to-nearest-even f32 -> bf16 bit pattern
  uint32_t u = __builtin_bit_cast(uint32_t, f);
  uint32_t r = (u + 0x7FFFu + ((u >> 16) & 1u)) >> 16;
  return (uint16_t)r;
}

// ---------------------------------------------------------------------------
// Bulk zero (halo init for padded NHWC buffers)
// ---------------------------------------------------------------------------
__global__ void zero_u4_kernel(uint4* __restrict__ p, size_t n16) {
  size_t i = (size_t)blockIdx.x * blockDim.x + threadIdx.x;
  size_t stride = (size_t)gridDim.x * blockDim.x;
  uint4 z; z.x = 0u; z.y = 0u; z.z = 0u; z.w = 0u;
  for (; i < n16; i += stride) p[i] = z;
}

// ---------------------------------------------------------------------------
// Weight repack: OIHW f32 (or [O][I] fc) -> bf16 B-fragment layout.
// Fragment f = (t*KC + kc)*OCG + g; within fragment: lane L holds column
// n = g*16 + (L&15), k = (L>>4)*8 + (e&7) + (e>>3)*16, 16 bf16 -> one 32B read.
// ---------------------------------------------------------------------------
__global__ void repack_w_kernel(const float* __restrict__ w, uint16_t* __restrict__ out,
                                int taps, int IC, int KC, int OCG, int total) {
  int idx = blockIdx.x * blockDim.x + threadIdx.x;
  if (idx >= total) return;
  int e    = idx & 15;
  int lane = (idx >> 4) & 31;
  int f    = idx >> 9;
  int g    = f % OCG;
  int kc   = (f / OCG) % KC;
  int t    = f / (OCG * KC);
  int n = g * 16 + (lane & 15);
  int k = kc * 32 + (lane >> 4) * 8 + (e & 7) + (e >> 3) * 16;
  float val = 0.0f;
  if (k < IC) val = w[((size_t)n * IC + k) * taps + t];
  out[idx] = f2bf(val);
}

// ---------------------------------------------------------------------------
// Pack conv1 input: concat(x_flow_warped, x_current, flow) NCHW f32
//   -> NHWC bf16 padded [B,258,258,160] (interior only; halo/extra ch zeroed)
// ---------------------------------------------------------------------------
__global__ void pack_omin_kernel(const float* __restrict__ xfw,
                                 const float* __restrict__ xc,
                                 const float* __restrict__ flow,
                                 uint16_t* __restrict__ dst) {
  int idx = blockIdx.x * blockDim.x + threadIdx.x; // B*H*W*130
  int c = idx % OMINC;
  int p = idx / OMINC;
  int w = p % NW; p /= NW;
  int h = p % NH;
  int b = p / NH;
  float v;
  if (c < NC)            v = xfw [((size_t)(b * NC + c)        * NH + h) * NW + w];
  else if (c < 2 * NC)   v = xc  [((size_t)(b * NC + (c - NC)) * NH + h) * NW + w];
  else                   v = flow[((size_t)(b * 2  + (c - 2 * NC)) * NH + h) * NW + w];
  dst[((size_t)(b * NHP + h + 1) * NWP + (w + 1)) * OMINCP + c] = f2bf(v);
}

// Pack x: NCHW f32 -> NHWC bf16 [B,256,256,64] (for vp WMMA GEMM)
__global__ void pack_x_kernel(const float* __restrict__ x, uint16_t* __restrict__ dst) {
  int idx = blockIdx.x * blockDim.x + threadIdx.x; // B*H*W*64
  int c = idx & 63;
  int p = idx >> 6;
  int w = p % NW; p /= NW;
  int h = p % NH;
  int b = p / NH;
  dst[((size_t)(b * NH + h) * NW + w) * NC + c] =
      f2bf(x[((size_t)(b * NC + c) * NH + h) * NW + w]);
}

// ---------------------------------------------------------------------------
// TDM: stage the packed weight blob (bytes, multiple of 8) into LDS.
// Builds a 2D D# (data_size=8B, tile = nunits x 1) per CDNA5 ISA.
// ---------------------------------------------------------------------------
#if USE_TDM
__device__ __forceinline__ void tdm_stage_weights(const uint16_t* wp,
                                                  uint32_t lds_off,
                                                  uint32_t bytes) {
  const uint64_t ga = (uint64_t)(uintptr_t)wp;
  const uint32_t nu = bytes >> 3;  // 8-byte units (fits 16-bit for all our blobs)
  u32x4 g0;
  g0[0] = 0x1u;                                      // count=1, user mode
  g0[1] = lds_off;                                   // lds_addr
  g0[2] = (uint32_t)ga;                              // global_addr[31:0]
  g0[3] = 0x80000000u | ((uint32_t)(ga >> 32) & 0x01FFFFFFu); // type=2 | addr[56:32]
  i32x8 g1;
  g1[0] = 0x00030000;                                // data_size=3 (8B), no flags
  g1[1] = (int)((nu & 0xFFFFu) << 16);               // tensor_dim0[15:0]
  g1[2] = (int)(((nu >> 16) & 0xFFFFu) | (1u << 16));// tensor_dim0[31:16], tensor_dim1=1
  g1[3] = (int)((nu & 0xFFFFu) << 16);               // tile_dim0 = nu
  g1[4] = 1;                                         // tile_dim1 = 1 (tile_dim2 unused)
  g1[5] = (int)nu;                                   // tensor_dim0_stride[31:0]
  g1[6] = 0;                                         // stride[47:32], dim1_stride lo
  g1[7] = 0;
  i32x4 gz4; gz4[0] = 0; gz4[1] = 0; gz4[2] = 0; gz4[3] = 0;
  i32x8 gz8;
#pragma unroll
  for (int i = 0; i < 8; ++i) gz8[i] = 0;
  __builtin_amdgcn_tensor_load_to_lds(g0, g1, gz4, gz4, gz8, 0);
  __builtin_amdgcn_s_wait_tensorcnt((short)0);
}
#endif

// ---------------------------------------------------------------------------
// Implicit-GEMM WMMA kernel. One wave computes 16 pixels x (OCG*16) channels.
// Weights staged in LDS (TDM) once per block; A loads software-pipelined.
//   SMODE : 0 = bf16 NHWC padded (+bias, leaky-relu)   [conv1..3]
//           1 = f32 NHWC unpadded (+bias)              [conv4 -> om]
//           2 = f32 NHWC padded   (+bias)              [vp -> vpad]
//           3 = f32 NCHW          (+bias)              [op -> d_out]
// ---------------------------------------------------------------------------
template <int TAPS, int KC, int OCG, int SMODE, bool INPAD>
__global__ __launch_bounds__(256) void gemm_wmma_kernel(
    const uint16_t* __restrict__ in, const uint16_t* __restrict__ wp,
    const float* __restrict__ bias, void* __restrict__ outp) {
  constexpr int ICP = KC * 32;
  constexpr int WBYTES = TAPS * KC * OCG * 512 * 2;
  extern __shared__ uint16_t smem[];

  const int lane = threadIdx.x & 31;
  const int wv   = threadIdx.x >> 5;

  // ---- stage packed weights into LDS (shared by all 8 waves) ----
#if USE_TDM
  if (wv == 0) {
    tdm_stage_weights(wp, (uint32_t)(uintptr_t)smem, WBYTES);
  }
#else
  for (int i = threadIdx.x; i < WBYTES / 16; i += 256) {
    ((uint4*)smem)[i] = ((const uint4*)wp)[i];
  }
#endif
  __syncthreads();

  const int tile = blockIdx.x * 8 + wv;        // grid sized exactly: EXEC all-1s
  const int WT   = NW / 16;
  const int b    = tile / (NH * WT);
  int rem        = tile % (NH * WT);
  const int h    = rem / WT;
  const int w0   = (rem % WT) * 16;
  const int mL   = lane & 15;                  // A-matrix row for this lane
  const int half = lane >> 4;                  // K-half selector
  constexpr int inHs = INPAD ? NHP : NH;
  constexpr int inWs = INPAD ? NWP : NW;

  v8f acc[OCG] = {};
  const uint16_t* bl = smem + lane * 16;

  // pipelined A-fragment loads
  const uint16_t* arow =
      in + ((size_t)(b * inHs + h) * inWs + (w0 + mL)) * ICP + (half ? 8 : 0);
  v8bf alo = *reinterpret_cast<const v8bf*>(arow);
  v8bf ahi = *reinterpret_cast<const v8bf*>(arow + 16);

  for (int t = 0; t < TAPS; ++t) {
#pragma unroll
    for (int kc = 0; kc < KC; ++kc) {
      v16bf a = __builtin_shufflevector(alo, ahi, 0, 1, 2, 3, 4, 5, 6, 7,
                                        8, 9, 10, 11, 12, 13, 14, 15);
      // issue next A loads before the WMMA group (overlap global latency)
      if (kc + 1 < KC) {
        alo = *reinterpret_cast<const v8bf*>(arow + (kc + 1) * 32);
        ahi = *reinterpret_cast<const v8bf*>(arow + (kc + 1) * 32 + 16);
      } else if (t + 1 < TAPS) {
        const int ndy = (TAPS == 9) ? ((t + 1) / 3) : 0;
        const int ndx = (TAPS == 9) ? ((t + 1) % 3) : 0;
        arow = in + ((size_t)(b * inHs + h + ndy) * inWs + (w0 + mL + ndx)) * ICP +
               (half ? 8 : 0);
        __builtin_prefetch(arow, 0, 3);
        alo = *reinterpret_cast<const v8bf*>(arow);
        ahi = *reinterpret_cast<const v8bf*>(arow + 16);
      }
      const uint16_t* bp = bl + (size_t)((t * KC + kc) * OCG) * 512;
#pragma unroll
      for (int g = 0; g < OCG; ++g) {
        v16bf bfrag = *reinterpret_cast<const v16bf*>(bp + (size_t)g * 512);
        acc[g] = __builtin_amdgcn_wmma_f32_16x16x32_bf16(
            false, a, false, bfrag, (short)0, acc[g], false, false);
      }
    }
  }

  // C/D layout: VGPR r -> pixel (half*8 + r), channel (lane&15) within group.
  const int n0 = mL;
#pragma unroll
  for (int g = 0; g < OCG; ++g) {
    const float bv = bias[g * 16 + n0];
#pragma unroll
    for (int r = 0; r < 8; ++r) {
      const int mo = half * 8 + r;
      float v = acc[g][r] + bv;
      if constexpr (SMODE == 0) {
        v = v >= 0.0f ? v : 0.1f * v;  // leaky relu
        uint16_t* out = (uint16_t*)outp;
        out[((size_t)(b * NHP + h + 1) * NWP + (w0 + mo + 1)) * (OCG * 16) +
            g * 16 + n0] = f2bf(v);
      } else if constexpr (SMODE == 1) {
        float* out = (float*)outp;
        out[((size_t)(b * NH + h) * NW + (w0 + mo)) * (OCG * 16) + g * 16 + n0] = v;
      } else if constexpr (SMODE == 2) {
        float* out = (float*)outp;
        out[((size_t)(b * NHP + h + 1) * NWP + (w0 + mo + 1)) * (OCG * 16) +
            g * 16 + n0] = v;
      } else {
        float* out = (float*)outp;
        out[((size_t)(b * NC + g * 16 + n0) * NH + h) * NW + (w0 + mo)] = v;
      }
    }
  }
}

// ---------------------------------------------------------------------------
// Deformable bilinear sampling. 1 thread = (b,h,w,g); 16 channels in regs.
// om: [B,256,256,112] f32 (per group: 18 offsets xy-interleaved, 9 masks)
// vpad: [B,258,258,64] f32 ; out: bf16 NHWC [B,256,256,64]
// ---------------------------------------------------------------------------
__global__ __launch_bounds__(256) void dcn_sample_kernel(
    const float* __restrict__ om, const float* __restrict__ vpad,
    uint16_t* __restrict__ outp) {
  int idx = blockIdx.x * blockDim.x + threadIdx.x; // B*H*W*G
  int g = idx & (NG - 1);
  int p = idx >> 2;
  int w = p % NW; p /= NW;
  int h = p % NH;
  int b = p / NH;
  const float* ob = om + ((size_t)(b * NH + h) * NW + w) * OMC + g * 27;

  float acc[NCG];
#pragma unroll
  for (int i = 0; i < NCG; ++i) acc[i] = 0.0f;

#pragma unroll
  for (int k = 0; k < 9; ++k) {
    const int dy = k / 3 - 1, dx = k % 3 - 1;
    const float offx = ob[2 * k];
    const float offy = ob[2 * k + 1];
    const float mk   = ob[18 + k];
    const float sy = (float)(h + 1 + dy) + offy;  // padded coords
    const float sx = (float)(w + 1 + dx) + offx;
    const float y0f = floorf(sy), x0f = floorf(sx);
    const float wy = sy - y0f, wx = sx - x0f;
    const int y0 = (int)y0f, x0 = (int)x0f;
    const float w00 = (1.f - wy) * (1.f - wx) * mk;
    const float w01 = (1.f - wy) * wx * mk;
    const float w10 = wy * (1.f - wx) * mk;
    const float w11 = wy * wx * mk;
#pragma unroll
    for (int cy = 0; cy < 2; ++cy) {
      const int yi = y0 + cy;
      const bool vy = (yi >= 0) && (yi < NHP);
      const int yc = yi < 0 ? 0 : (yi > NHP - 1 ? NHP - 1 : yi);
#pragma unroll
      for (int cx = 0; cx < 2; ++cx) {
        const int xi = x0 + cx;
        const bool vx = (xi >= 0) && (xi < NWP);
        const int xc = xi < 0 ? 0 : (xi > NWP - 1 ? NWP - 1 : xi);
        float wc = cy ? (cx ? w11 : w10) : (cx ? w01 : w00);
        wc = (vy && vx) ? wc : 0.0f;
        const float4* vp = reinterpret_cast<const float4*>(
            vpad + ((size_t)(b * NHP + yc) * NWP + xc) * NC + g * NCG);
#pragma unroll
        for (int q = 0; q < 4; ++q) {
          float4 vv = vp[q];
          acc[4 * q + 0] += wc * vv.x;
          acc[4 * q + 1] += wc * vv.y;
          acc[4 * q + 2] += wc * vv.z;
          acc[4 * q + 3] += wc * vv.w;
        }
      }
    }
  }
  uint16_t* o = outp + ((size_t)(b * NH + h) * NW + w) * NC + g * NCG;
#pragma unroll
  for (int i = 0; i < NCG; ++i) o[i] = f2bf(acc[i]);
}

// ---------------------------------------------------------------------------
// Host side
// ---------------------------------------------------------------------------
static inline size_t al256(size_t x) { return (x + 255) & ~(size_t)255; }

extern "C" void kernel_launch(void* const* d_in, const int* in_sizes, int n_in,
                              void* d_out, int out_size, void* d_ws, size_t ws_size,
                              hipStream_t stream) {
  (void)in_sizes; (void)n_in; (void)out_size; (void)ws_size;
  const float* x    = (const float*)d_in[0];
  const float* xfw  = (const float*)d_in[1];
  const float* xc   = (const float*)d_in[2];
  const float* flow = (const float*)d_in[3];
  const float* w1   = (const float*)d_in[4];
  const float* b1   = (const float*)d_in[5];
  const float* w2   = (const float*)d_in[6];
  const float* b2   = (const float*)d_in[7];
  const float* w3   = (const float*)d_in[8];
  const float* b3   = (const float*)d_in[9];
  const float* w4   = (const float*)d_in[10];
  const float* b4   = (const float*)d_in[11];
  const float* vpw  = (const float*)d_in[12];
  const float* vpb  = (const float*)d_in[13];
  const float* opw  = (const float*)d_in[14];
  const float* opb  = (const float*)d_in[15];
  float* out = (float*)d_out;

  // workspace layout
  char* ws = (char*)d_ws;
  size_t off = 0;
  const size_t sz_omin = (size_t)NB * NHP * NWP * OMINCP * 2;   // bf16 padded
  const size_t sz_hbuf = (size_t)NB * NHP * NWP * NC * 2;       // bf16 padded
  const size_t sz_vpad = (size_t)NB * NHP * NWP * NC * 4;       // f32 padded
  const size_t sz_om   = (size_t)NB * NH * NW * OMC * 4;        // f32
  const size_t sz_xw   = (size_t)NB * NH * NW * NC * 2;         // bf16
  const size_t sz_dcn  = (size_t)NB * NH * NW * NC * 2;         // bf16
  uint16_t* om_in = (uint16_t*)(ws + off); off += al256(sz_omin);
  uint16_t* hA    = (uint16_t*)(ws + off); off += al256(sz_hbuf);
  uint16_t* hB    = (uint16_t*)(ws + off); off += al256(sz_hbuf);
  float*    vpad  = (float*)   (ws + off); off += al256(sz_vpad);
  float*    om    = (float*)   (ws + off); off += al256(sz_om);
  uint16_t* xw    = (uint16_t*)(ws + off); off += al256(sz_xw);
  uint16_t* dcn   = (uint16_t*)(ws + off); off += al256(sz_dcn);
  // packed weights (fragments of 512 bf16 each)
  const int kc1 = OMINCP / 32, kcc = NC / 32;
  const int nw1 = 9 * kc1 * 4 * 512;
  const int nwc = 9 * kcc * 4 * 512;
  const int nw4 = 9 * kcc * (OMC / 16) * 512;
  const int nfc = 1 * kcc * 4 * 512;
  uint16_t* w1p = (uint16_t*)(ws + off); off += al256((size_t)nw1 * 2);
  uint16_t* w2p = (uint16_t*)(ws + off); off += al256((size_t)nwc * 2);
  uint16_t* w3p = (uint16_t*)(ws + off); off += al256((size_t)nwc * 2);
  uint16_t* w4p = (uint16_t*)(ws + off); off += al256((size_t)nw4 * 2);
  uint16_t* vpp = (uint16_t*)(ws + off); off += al256((size_t)nfc * 2);
  uint16_t* opp = (uint16_t*)(ws + off); off += al256((size_t)nfc * 2);

  // 1) zero padded buffers (halo correctness)
  zero_u4_kernel<<<2048, 256, 0, stream>>>((uint4*)om_in, sz_omin / 16);
  zero_u4_kernel<<<2048, 256, 0, stream>>>((uint4*)hA,    sz_hbuf / 16);
  zero_u4_kernel<<<2048, 256, 0, stream>>>((uint4*)hB,    sz_hbuf / 16);
  zero_u4_kernel<<<2048, 256, 0, stream>>>((uint4*)vpad,  sz_vpad / 16);

  // 2) repack weights into WMMA B-fragment layout
  repack_w_kernel<<<(nw1 + 255) / 256, 256, 0, stream>>>(w1, w1p, 9, OMINC, kc1, 4, nw1);
  repack_w_kernel<<<(nwc + 255) / 256, 256, 0, stream>>>(w2, w2p, 9, NC, kcc, 4, nwc);
  repack_w_kernel<<<(nwc + 255) / 256, 256, 0, stream>>>(w3, w3p, 9, NC, kcc, 4, nwc);
  repack_w_kernel<<<(nw4 + 255) / 256, 256, 0, stream>>>(w4, w4p, 9, NC, kcc, OMC / 16, nw4);
  repack_w_kernel<<<(nfc + 255) / 256, 256, 0, stream>>>(vpw, vpp, 1, NC, kcc, 4, nfc);
  repack_w_kernel<<<(nfc + 255) / 256, 256, 0, stream>>>(opw, opp, 1, NC, kcc, 4, nfc);

  // 3) pack activations (NCHW f32 -> NHWC bf16)
  {
    int tot = NB * NH * NW * OMINC;
    pack_omin_kernel<<<tot / 256, 256, 0, stream>>>(xfw, xc, flow, om_in);
  }
  {
    int tot = NB * NH * NW * NC;
    pack_x_kernel<<<tot / 256, 256, 0, stream>>>(x, xw);
  }

  // 4) conv stack (implicit GEMM, bf16 WMMA, weights staged in LDS via TDM)
  const int conv_blocks = (NB * NH * (NW / 16)) / 8;   // 2048, exact
  const size_t lds1 = (size_t)9 * kc1 * 4 * 1024;      // 184320 B
  const size_t ldsc = (size_t)9 * kcc * 4 * 1024;      // 73728 B
  const size_t lds4 = (size_t)9 * kcc * (OMC / 16) * 1024; // 129024 B
  const size_t ldsf = (size_t)1 * kcc * 4 * 1024;      // 8192 B
  gemm_wmma_kernel<9, 5, 4, 0, true><<<conv_blocks, 256, lds1, stream>>>(om_in, w1p, b1, hA);
  gemm_wmma_kernel<9, 2, 4, 0, true><<<conv_blocks, 256, ldsc, stream>>>(hA, w2p, b2, hB);
  gemm_wmma_kernel<9, 2, 4, 0, true><<<conv_blocks, 256, ldsc, stream>>>(hB, w3p, b3, hA);
  gemm_wmma_kernel<9, 2, 7, 1, true><<<conv_blocks, 256, lds4, stream>>>(hA, w4p, b4, om);

  // 5) value projection: v = x @ vp_w^T + vp_b, written into padded buffer
  const int fc_blocks = (NB * NH * NW / 16) / 8;       // 2048, exact
  gemm_wmma_kernel<1, 2, 4, 2, false><<<fc_blocks, 256, ldsf, stream>>>(xw, vpp, vpb, vpad);

  // 6) deformable bilinear sampling
  {
    int tot = NB * NH * NW * NG;
    dcn_sample_kernel<<<tot / 256, 256, 0, stream>>>(om, vpad, dcn);
  }

  // 7) output projection -> NCHW f32 d_out
  gemm_wmma_kernel<1, 2, 4, 3, false><<<fc_blocks, 256, ldsf, stream>>>(dcn, opp, opb, out);
}
